// VecAwCell_9337258901888
// MI455X (gfx1250) — compile-verified, matrix operand order checked by reading, not run on hardware
//
#include <hip/hip_runtime.h>
#include <hip/hip_bf16.h>

#define UNITS 512
#define BATCH 64
#define TSEQ 512
#define FDIM 512
#define NHEADS 8
#define HDIM 64
#define NWG 16
#define NTHREADS 256
#define WMAT_SZ (UNITS * UNITS)

typedef __bf16 bf16_t;
typedef bf16_t v16bf __attribute__((ext_vector_type(16)));
typedef float  v8f   __attribute__((ext_vector_type(8)));

// ---------- helpers ----------

__device__ __forceinline__ bf16_t f2bf(float f) {
  union { float f; unsigned u; } a; a.f = f;
  unsigned r = a.u + 0x7FFFu + ((a.u >> 16) & 1u);   // round-to-nearest-even
  union { unsigned short s; bf16_t h; } b; b.s = (unsigned short)(r >> 16);
  return b.h;
}

__device__ __forceinline__ float sig(float x) { return 1.0f / (1.0f + __expf(-x)); }

// Generic bf16 fragment load (A or B side), row-major, leading dim in elements.
// Lane L holds row/col = L&15 ; K = kh..kh+7 and kh+16..kh+23, kh = (L>=16)*8.
__device__ __forceinline__ v16bf loadFrag_bf16(const bf16_t* __restrict__ base, int ld,
                                               int k0) {
  const int lane = threadIdx.x & 31;
  const int r_   = lane & 15;
  const int kh   = (lane >> 4) << 3;
  const bf16_t* p = base + r_ * ld + kh + k0;
  union { uint4 u[2]; v16bf v; } r;
  r.u[0] = *reinterpret_cast<const uint4*>(p);
  r.u[1] = *reinterpret_cast<const uint4*>(p + 16);
  return r.v;
}

// A-fragment converted on the fly from f32 (fallback when workspace can't hold xbf).
__device__ __forceinline__ v16bf loadA_f32(const float* __restrict__ base, int lda, int k0) {
  const int lane = threadIdx.x & 31;
  const int m  = lane & 15;
  const int kh = (lane >> 4) << 3;
  const float* p = base + m * lda + kh + k0;
  float4 x0 = *reinterpret_cast<const float4*>(p);
  float4 x1 = *reinterpret_cast<const float4*>(p + 4);
  float4 x2 = *reinterpret_cast<const float4*>(p + 16);
  float4 x3 = *reinterpret_cast<const float4*>(p + 20);
  v16bf a;
  a[0] = f2bf(x0.x); a[1] = f2bf(x0.y); a[2]  = f2bf(x0.z); a[3]  = f2bf(x0.w);
  a[4] = f2bf(x1.x); a[5] = f2bf(x1.y); a[6]  = f2bf(x1.z); a[7]  = f2bf(x1.w);
  a[8] = f2bf(x2.x); a[9] = f2bf(x2.y); a[10] = f2bf(x2.z); a[11] = f2bf(x2.w);
  a[12]= f2bf(x3.x); a[13]= f2bf(x3.y); a[14] = f2bf(x3.z); a[15] = f2bf(x3.w);
  return a;
}

// One 16x16 tile, A (arbitrary ld) and B both bf16 (pure b128 loads + WMMA).
__device__ __forceinline__ v8f mm512_bf_ld(const bf16_t* __restrict__ Abase, int lda,
                                           const bf16_t* __restrict__ WtCol, v8f acc) {
#pragma unroll 4
  for (int k0 = 0; k0 < UNITS; k0 += 32) {
    v16bf a = loadFrag_bf16(Abase, lda, k0);
    v16bf b = loadFrag_bf16(WtCol, UNITS, k0);
    acc = __builtin_amdgcn_wmma_f32_16x16x32_bf16(false, a, false, b, (short)0, acc,
                                                  false, false);
  }
  return acc;
}

__device__ __forceinline__ v8f mm512_bf(const bf16_t* __restrict__ Abase,
                                        const bf16_t* __restrict__ WtCol, v8f acc) {
  return mm512_bf_ld(Abase, UNITS, WtCol, acc);
}

// One 16x16 tile, A f32 with on-the-fly convert (x fallback only).
__device__ __forceinline__ v8f mm512_f32A(const float* __restrict__ Abase, int lda,
                                          const bf16_t* __restrict__ WtCol, v8f acc) {
#pragma unroll 1
  for (int k0 = 0; k0 < UNITS; k0 += 32) {
    v16bf a = loadA_f32(Abase, lda, k0);
    v16bf b = loadFrag_bf16(WtCol, UNITS, k0);
    acc = __builtin_amdgcn_wmma_f32_16x16x32_bf16(false, a, false, b, (short)0, acc,
                                                  false, false);
  }
  return acc;
}

__device__ __forceinline__ void store_tile_f32(float* __restrict__ C, int rowBase,
                                               int colBase, v8f acc) {
  const int lane = threadIdx.x & 31;
  const int n  = lane & 15;
  const int mo = (lane >> 4) << 3;
#pragma unroll
  for (int r = 0; r < 8; ++r)
    C[(rowBase + mo + r) * UNITS + colBase + n] = acc[r];
}

__device__ __forceinline__ void store_tile_bf16(bf16_t* __restrict__ C, int rowBase,
                                                int colBase, v8f acc) {
  const int lane = threadIdx.x & 31;
  const int n  = lane & 15;
  const int mo = (lane >> 4) << 3;
#pragma unroll
  for (int r = 0; r < 8; ++r)
    C[(rowBase + mo + r) * UNITS + colBase + n] = f2bf(acc[r]);
}

// ---------- persistent kernel ----------

__global__ __launch_bounds__(NTHREADS, 1)
void lstm_attn_persistent(
    const float* __restrict__ x,
    const float* __restrict__ kx, const float* __restrict__ kh_w,
    const float* __restrict__ wq, const float* __restrict__ wk,
    const float* __restrict__ wvm, const float* __restrict__ wo,
    const float* __restrict__ wf, const float* __restrict__ wi,
    const float* __restrict__ g1, const float* __restrict__ b1,
    const float* __restrict__ g2, const float* __restrict__ b2,
    float* __restrict__ dout,
    unsigned* __restrict__ syncc,
    float* __restrict__ xp,  float* __restrict__ zf, float* __restrict__ zi,
    float* __restrict__ shn, float* __restrict__ c,
    float* __restrict__ q,   float* __restrict__ k,  float* __restrict__ v,
    bf16_t* __restrict__ hb,  bf16_t* __restrict__ xpb, bf16_t* __restrict__ shb,
    bf16_t* __restrict__ lnb, bf16_t* __restrict__ mdb, bf16_t* __restrict__ avb,
    bf16_t* __restrict__ wt,  bf16_t* __restrict__ xbf /* nullptr => f32 fallback */) {

  const int tid   = blockIdx.x * NTHREADS + threadIdx.x;
  const int wvid  = threadIdx.x >> 5;                       // wave in block (0..7)
  const int gw    = blockIdx.x * (NTHREADS / 32) + wvid;    // global wave (0..127)
  const int lane  = threadIdx.x & 31;

  __shared__ float sred[4][64][4];
  __shared__ float sbc[4][4];
  __shared__ float ssc[8][64];

  unsigned bgen = 0;
  auto gbar = [&]() {
    __syncthreads();
    ++bgen;
    if (threadIdx.x == 0) {
      __threadfence();
      atomicAdd(syncc, 1u);
      const unsigned tgt = bgen * NWG;
      while (*((volatile unsigned*)syncc) < tgt) __builtin_amdgcn_s_sleep(2);
    }
    __syncthreads();
    __threadfence();
  };

  // ---- phase 0: weights -> bf16 transposed Wt[n][k]; x -> bf16; zero hb, c ----
  {
    const float* wsrc[8] = {kx, kh_w, wf, wi, wq, wk, wvm, wo};
#pragma unroll 1
    for (int w = 0; w < 8; ++w) {
      const float* s = wsrc[w];
      bf16_t* d = wt + w * WMAT_SZ;
      for (int i = tid; i < WMAT_SZ; i += NWG * NTHREADS) {
        int n = i >> 9, kk = i & 511;
        d[i] = f2bf(s[kk * UNITS + n]);
      }
    }
    if (xbf) {
      const int NTOT = BATCH * TSEQ * FDIM;
      for (int i = tid; i < NTOT; i += NWG * NTHREADS) xbf[i] = f2bf(x[i]);
    }
    const bf16_t z = f2bf(0.0f);
    for (int i = tid; i < BATCH * UNITS; i += NWG * NTHREADS) { hb[i] = z; c[i] = 0.f; }
  }
  gbar();

  const int tm = gw & 3, tn = gw >> 2;        // 4 x 32 tiles of [64 x 512]
  const int rowBase = tm * 16, colBase = tn * 16;
  const int cn = lane & 15, cmo = (lane >> 4) << 3;

#pragma unroll 1
  for (int t = 0; t < TSEQ; ++t) {
    // ---- P1: xp = x_t @ Kx + h @ Kh ----
    {
      v8f acc = {};
      if (xbf) {
        acc = mm512_bf_ld(xbf + (size_t)rowBase * (TSEQ * FDIM) + (size_t)t * FDIM,
                          TSEQ * FDIM, wt + 0 * WMAT_SZ + colBase * UNITS, acc);
      } else {
        acc = mm512_f32A(x + (size_t)rowBase * (TSEQ * FDIM) + (size_t)t * FDIM,
                         TSEQ * FDIM, wt + 0 * WMAT_SZ + colBase * UNITS, acc);
      }
      acc = mm512_bf(hb + rowBase * UNITS, wt + 1 * WMAT_SZ + colBase * UNITS, acc);
      store_tile_f32(xp, rowBase, colBase, acc);
      store_tile_bf16(xpb, rowBase, colBase, acc);
    }
    gbar();

    // ---- P2: zf = xp @ Wf ; zi = xp @ Wi ----
    {
      v8f a = mm512_bf(xpb + rowBase * UNITS, wt + 2 * WMAT_SZ + colBase * UNITS, v8f{});
      store_tile_f32(zf, rowBase, colBase, a);
      v8f bacc = mm512_bf(xpb + rowBase * UNITS, wt + 3 * WMAT_SZ + colBase * UNITS, v8f{});
      store_tile_f32(zi, rowBase, colBase, bacc);
    }
    gbar();

    // ---- P3: gates, cell update, silu, two LayerNorms ----
    {
      const int lrow = threadIdx.x >> 6;      // 4 rows per workgroup
      const int tIn  = threadIdx.x & 63;      // 64 threads per row
      const int row  = blockIdx.x * 4 + lrow; // 0..63
      float lt[8], sh[8];
      float s_lt = 0.f, q_lt = 0.f, s_sh = 0.f, q_sh = 0.f;
#pragma unroll
      for (int e = 0; e < 8; ++e) {
        int idx = row * UNITS + e * 64 + tIn;
        float xpv = xp[idx];
        float fg = sig(zf[idx]);
        float ig = sig(zi[idx]);
        float cand = tanhf(xpv);
        float ltv = fg * c[idx] + ig * cand;
        float mv = xpv * sig(xpv);              // silu(xp)
        float sv = ltv + mv;
        sv = sv * sig(sv);                      // silu(longterm + medium)
        lt[e] = ltv; sh[e] = sv;
        c[idx] = ltv; mdb[idx] = f2bf(mv);
        s_lt += ltv; q_lt += ltv * ltv; s_sh += sv; q_sh += sv * sv;
      }
      sred[lrow][tIn][0] = s_lt; sred[lrow][tIn][1] = q_lt;
      sred[lrow][tIn][2] = s_sh; sred[lrow][tIn][3] = q_sh;
      __syncthreads();
      if (tIn == 0) {
        float a0 = 0.f, a1 = 0.f, a2 = 0.f, a3 = 0.f;
        for (int j = 0; j < 64; ++j) {
          a0 += sred[lrow][j][0]; a1 += sred[lrow][j][1];
          a2 += sred[lrow][j][2]; a3 += sred[lrow][j][3];
        }
        float mu_l = a0 * (1.0f / 512.f), var_l = a1 * (1.0f / 512.f) - mu_l * mu_l;
        float mu_s = a2 * (1.0f / 512.f), var_s = a3 * (1.0f / 512.f) - mu_s * mu_s;
        sbc[lrow][0] = mu_l; sbc[lrow][1] = rsqrtf(var_l + 1e-3f);
        sbc[lrow][2] = mu_s; sbc[lrow][3] = rsqrtf(var_s + 1e-3f);
      }
      __syncthreads();
      const float mu_l = sbc[lrow][0], rs_l = sbc[lrow][1];
      const float mu_s = sbc[lrow][2], rs_s = sbc[lrow][3];
#pragma unroll
      for (int e = 0; e < 8; ++e) {
        int col = e * 64 + tIn, idx = row * UNITS + col;
        float sv = (sh[e] - mu_s) * rs_s * g1[col] + b1[col];
        float lv = (lt[e] - mu_l) * rs_l * g2[col] + b2[col];
        shn[idx] = sv;
        shb[idx] = f2bf(sv);
        lnb[idx] = f2bf(lv);
      }
    }
    gbar();

    // ---- P4: q = shn@Wq ; k = lnn@Wk ; v = med@Wv ----
    {
      v8f a = mm512_bf(shb + rowBase * UNITS, wt + 4 * WMAT_SZ + colBase * UNITS, v8f{});
      store_tile_f32(q, rowBase, colBase, a);
      v8f bacc = mm512_bf(lnb + rowBase * UNITS, wt + 5 * WMAT_SZ + colBase * UNITS, v8f{});
      store_tile_f32(k, rowBase, colBase, bacc);
      v8f cacc = mm512_bf(mdb + rowBase * UNITS, wt + 6 * WMAT_SZ + colBase * UNITS, v8f{});
      store_tile_f32(v, rowBase, colBase, cacc);
    }
    gbar();

    // ---- P5: per-batch 8x8 head attention (one wave per batch) ----
    {
      const int bidx = gw;
      if (bidx < BATCH) {
        const float* qb = q + bidx * UNITS;
        const float* kb = k + bidx * UNITS;
#pragma unroll
        for (int rep = 0; rep < 2; ++rep) {
          int sidx = lane + rep * 32;
          int hh = sidx >> 3, gg = sidx & 7;
          float s = 0.f;
          for (int d = 0; d < HDIM; ++d) s += qb[hh * HDIM + d] * kb[gg * HDIM + d];
          ssc[wvid][sidx] = s * 0.125f;     // 1/sqrt(64)
        }
      }
      __syncthreads();
      if (bidx < BATCH && lane < 8) {
        float m = -1e30f;
        for (int gg = 0; gg < 8; ++gg) m = fmaxf(m, ssc[wvid][lane * 8 + gg]);
        float sum = 0.f, e[8];
        for (int gg = 0; gg < 8; ++gg) { e[gg] = __expf(ssc[wvid][lane * 8 + gg] - m); sum += e[gg]; }
        float inv = 1.0f / sum;
        for (int gg = 0; gg < 8; ++gg) ssc[wvid][lane * 8 + gg] = e[gg] * inv;
      }
      __syncthreads();
      if (bidx < BATCH) {
        const float* vb = v + bidx * UNITS;
#pragma unroll 4
        for (int j = 0; j < 16; ++j) {
          int idx = lane + j * 32;
          int hh = idx >> 6, dd = idx & 63;
          float s = 0.f;
#pragma unroll
          for (int gg = 0; gg < 8; ++gg) s += ssc[wvid][hh * 8 + gg] * vb[gg * HDIM + dd];
          avb[bidx * UNITS + idx] = f2bf(s);
        }
      }
    }
    gbar();

    // ---- P6: out = shn + attv @ Wo ; h = out (bf16) ; dout[:, t, :] = out ----
    {
      v8f a = mm512_bf(avb + rowBase * UNITS, wt + 7 * WMAT_SZ + colBase * UNITS, v8f{});
#pragma unroll
      for (int r = 0; r < 8; ++r) {
        int row = rowBase + cmo + r, col = colBase + cn;
        float o = shn[row * UNITS + col] + a[r];
        hb[row * UNITS + col] = f2bf(o);
        dout[(size_t)row * (TSEQ * UNITS) + (size_t)t * UNITS + col] = o;
      }
    }
    gbar();
  }
}

// ---------- host launcher ----------

extern "C" void kernel_launch(void* const* d_in, const int* in_sizes, int n_in,
                              void* d_out, int out_size, void* d_ws, size_t ws_size,
                              hipStream_t stream) {
  (void)in_sizes; (void)n_in; (void)out_size;
  const float* x  = (const float*)d_in[0];
  const float* kx = (const float*)d_in[1];
  const float* kh = (const float*)d_in[2];
  const float* wq = (const float*)d_in[3];
  const float* wk = (const float*)d_in[4];
  const float* wv = (const float*)d_in[5];
  const float* wo = (const float*)d_in[6];
  const float* wf = (const float*)d_in[7];
  const float* wi = (const float*)d_in[8];
  const float* g1 = (const float*)d_in[9];
  const float* b1 = (const float*)d_in[10];
  const float* g2 = (const float*)d_in[11];
  const float* b2 = (const float*)d_in[12];

  char* ws = (char*)d_ws;
  size_t off = 0;
  auto alloc = [&](size_t bytes) -> char* {
    char* p = ws + off;
    off += (bytes + 255) & ~(size_t)255;
    return p;
  };
  unsigned* syncc = (unsigned*)alloc(256);
  float* xp   = (float*)alloc((size_t)BATCH * UNITS * 4);
  float* zf   = (float*)alloc((size_t)BATCH * UNITS * 4);
  float* zi   = (float*)alloc((size_t)BATCH * UNITS * 4);
  float* shn  = (float*)alloc((size_t)BATCH * UNITS * 4);
  float* c    = (float*)alloc((size_t)BATCH * UNITS * 4);
  float* q    = (float*)alloc((size_t)BATCH * UNITS * 4);
  float* k    = (float*)alloc((size_t)BATCH * UNITS * 4);
  float* v    = (float*)alloc((size_t)BATCH * UNITS * 4);
  bf16_t* hb  = (bf16_t*)alloc((size_t)BATCH * UNITS * 2);
  bf16_t* xpb = (bf16_t*)alloc((size_t)BATCH * UNITS * 2);
  bf16_t* shb = (bf16_t*)alloc((size_t)BATCH * UNITS * 2);
  bf16_t* lnb = (bf16_t*)alloc((size_t)BATCH * UNITS * 2);
  bf16_t* mdb = (bf16_t*)alloc((size_t)BATCH * UNITS * 2);
  bf16_t* avb = (bf16_t*)alloc((size_t)BATCH * UNITS * 2);
  bf16_t* wt  = (bf16_t*)alloc((size_t)8 * WMAT_SZ * 2);

  // Optional bf16 copy of the entire x tensor (32 MB): use it only if ws has room.
  const size_t xbf_bytes = (size_t)BATCH * TSEQ * FDIM * 2;
  bf16_t* xbf = nullptr;
  if (off + xbf_bytes <= ws_size) xbf = (bf16_t*)alloc(xbf_bytes);

  (void)hipMemsetAsync(syncc, 0, 256, stream);
  hipLaunchKernelGGL(lstm_attn_persistent, dim3(NWG), dim3(NTHREADS), 0, stream,
                     x, kx, kh, wq, wk, wv, wo, wf, wi, g1, b1, g2, b2,
                     (float*)d_out, syncc,
                     xp, zf, zi, shn, c, q, k, v,
                     hb, xpb, shb, lnb, mdb, avb, wt, xbf);
}
